// MultiHeadAttention3D_51453708206669
// MI455X (gfx1250) — matrix operand order repaired and checked by reading, not compiled, not run on hardware
//
#include <hip/hip_runtime.h>
#include <hip/hip_bf16.h>

typedef __attribute__((ext_vector_type(16))) _Float16 v16h;
typedef __attribute__((ext_vector_type(8)))  _Float16 v8h;
typedef __attribute__((ext_vector_type(8)))  float    v8f;

#define DIM 512
#define NHEADS 16
#define HD 32
#define BATCH 2
#define TLEN 1728
#define MROWS (BATCH*TLEN)   // 3456
#define MTILES (MROWS/16)    // 216
#define STILES (TLEN/16)     // 108
#define EPSN 1e-12f

static __device__ __forceinline__ v16h combine16(v8h a, v8h b) {
  v16h r;
#pragma unroll
  for (int i = 0; i < 8; ++i) { r[i] = a[i]; r[i + 8] = b[i]; }
  return r;
}

// B fragment (16-bit, K=32 x N=16). Source: row-major [N x stride] f16 array,
// i.e. column n's K-values are contiguous. Lane l<16 holds col n0+l, K=k0..k0+7, k0+16..23;
// lanes 16-31 hold K=k0+8..15, k0+24..31 (matches A fragment K order).
static __device__ __forceinline__ v16h loadB(const _Float16* base, int stride,
                                             int l, int half, int n0, int k0) {
  const _Float16* p = base + (size_t)(n0 + l) * stride + k0 + half * 8;
  v8h a = *(const v8h*)p;
  v8h b = *(const v8h*)(p + 16);
  return combine16(a, b);
}

// A fragment (16-bit, M=16 x K=32) from row-major [M x stride] f16 array.
// Lane l<16: row m0+l, K = k0+0..7 then k0+16..23; lanes 16-31: K = k0+8..15, k0+24..31.
static __device__ __forceinline__ v16h loadA_h(const _Float16* base, int stride,
                                               int l, int half, int m0, int k0) {
  const _Float16* p = base + (size_t)(m0 + l) * stride + k0 + half * 8;
  v8h a = *(const v8h*)p;
  v8h b = *(const v8h*)(p + 16);
  return combine16(a, b);
}

// A fragment with on-the-fly fp32 -> f16 conversion.
static __device__ __forceinline__ v16h loadA_f32(const float* base, int stride,
                                                 int l, int half, int m0, int k0) {
  const float* r = base + (size_t)(m0 + l) * stride + k0 + half * 8;
  v16h f;
#pragma unroll
  for (int j = 0; j < 8; ++j) f[j] = (_Float16)r[j];
#pragma unroll
  for (int j = 0; j < 8; ++j) f[j + 8] = (_Float16)r[16 + j];
  return f;
}

// Reductions across the 16-lane half-groups (C/D layout: one row spans lanes 0-15 or 16-31).
static __device__ __forceinline__ float rowmax16(float x) {
#pragma unroll
  for (int m = 1; m < 16; m <<= 1) x = fmaxf(x, __shfl_xor(x, m, 32));
  return x;
}
static __device__ __forceinline__ float rowsum16(float x) {
#pragma unroll
  for (int m = 1; m < 16; m <<= 1) x += __shfl_xor(x, m, 32);
  return x;
}

// ---------------- weight fp32 -> f16 ----------------
__global__ __launch_bounds__(256) void cvt_w_kernel(const float* __restrict__ a,
                                                    const float* __restrict__ b,
                                                    const float* __restrict__ c,
                                                    const float* __restrict__ d,
                                                    _Float16* __restrict__ out) {
  int i = blockIdx.x * 256 + threadIdx.x;          // 0 .. 4*512*512-1
  const float* src = (i < 262144) ? a : (i < 524288) ? b : (i < 786432) ? c : d;
  out[i] = (_Float16)src[i & 262143];
}

// ---------------- fused QKV projection + cosine normalize ----------------
// One wave computes a 16(t) x 32(= one head's full dim) tile: out = x @ W^T + bias.
__global__ __launch_bounds__(256) void qkv_proj_kernel(
    const float* __restrict__ xq, const float* __restrict__ xk, const float* __restrict__ xv,
    const _Float16* __restrict__ WH,
    const float* __restrict__ bq, const float* __restrict__ bk, const float* __restrict__ bv,
    const float* __restrict__ logit_scale,
    _Float16* __restrict__ qh, _Float16* __restrict__ kh, _Float16* __restrict__ vT) {
  const int lane = threadIdx.x & 31;
  const int wid  = threadIdx.x >> 5;
  const int l = lane & 15, half = lane >> 4;
  const int wl = blockIdx.x * 8 + wid;             // 0 .. 3*216*16-1
  const int mid = wl / (MTILES * NHEADS);
  const int r   = wl - mid * MTILES * NHEADS;
  const int mt  = r >> 4;
  const int h   = r & 15;
  const int m0  = mt * 16;
  const int n0  = h * HD;

  const float*    x    = (mid == 0) ? xq : (mid == 1) ? xk : xv;
  const _Float16* W    = WH + (size_t)mid * DIM * DIM;   // [N=512][K=512] row-major
  const float*    bias = (mid == 0) ? bq : (mid == 1) ? bk : bv;

  v8f c0 = {}; v8f c1 = {};
#pragma unroll 4
  for (int k0 = 0; k0 < DIM; k0 += 32) {
    v16h a  = loadA_f32(x, DIM, l, half, m0, k0);
    v16h b0 = loadB(W, DIM, l, half, n0,      k0);
    v16h b1 = loadB(W, DIM, l, half, n0 + 16, k0);
    c0 = __builtin_amdgcn_wmma_f32_16x16x32_f16(false, a, false, b0, (short)0, c0, false, false);
    c1 = __builtin_amdgcn_wmma_f32_16x16x32_f16(false, a, false, b1, (short)0, c1, false, false);
  }
  const float bv0 = bias[n0 + l], bv1 = bias[n0 + 16 + l];
#pragma unroll
  for (int i = 0; i < 8; ++i) { c0[i] += bv0; c1[i] += bv1; }

  if (mid < 2) {
    const float ls = (mid == 0) ? logit_scale[0] : 1.0f;
#pragma unroll
    for (int i = 0; i < 8; ++i) {
      float s = rowsum16(c0[i] * c0[i] + c1[i] * c1[i]);   // per-row L2 over all 32 dims
      float scale = ls / fmaxf(sqrtf(s), EPSN);
      c0[i] *= scale; c1[i] *= scale;
    }
  }

  const int b  = m0 / TLEN;
  const int t0 = m0 - b * TLEN;
  if (mid < 2) {
    _Float16* dst = ((mid == 0) ? qh : kh) + (size_t)(b * NHEADS + h) * TLEN * HD;
#pragma unroll
    for (int i = 0; i < 8; ++i) {
      int t = t0 + i + 8 * half;
      dst[(size_t)t * HD + l]      = (_Float16)c0[i];
      dst[(size_t)t * HD + 16 + l] = (_Float16)c1[i];
    }
  } else {
    // V stored transposed [b,h,d,t] so attn B-fragments are contiguous per-lane.
    _Float16* dst = vT + (size_t)(b * NHEADS + h) * HD * TLEN;
#pragma unroll
    for (int i = 0; i < 8; ++i) {
      int t = t0 + i + 8 * half;
      dst[(size_t)l * TLEN + t]        = (_Float16)c0[i];
      dst[(size_t)(16 + l) * TLEN + t] = (_Float16)c1[i];
    }
  }
}

// ---------------- flash attention core ----------------
__global__ __launch_bounds__(256) void attn_kernel(
    const _Float16* __restrict__ qh, const _Float16* __restrict__ kh,
    const _Float16* __restrict__ vT, const float* __restrict__ rel_bias,
    _Float16* __restrict__ ctxh) {
  __shared__ __align__(16) _Float16 psh[8][16 * 32];   // 1KB per wave: P tile reshuffle
  const int lane = threadIdx.x & 31;
  const int wid  = threadIdx.x >> 5;
  const int l = lane & 15, half = lane >> 4;
  const int bh = blockIdx.y;                 // 0..31
  const int b = bh >> 4, h = bh & 15;
  const int qt = blockIdx.x * 8 + wid;
  if (qt >= STILES) return;                  // wave-uniform; EXEC stays all-1s for WMMA
  const int t0 = qt * 16;

  const _Float16* qp = qh + (size_t)(b * NHEADS + h) * TLEN * HD;
  const _Float16* kp = kh + (size_t)(b * NHEADS + h) * TLEN * HD;
  const _Float16* vp = vT + (size_t)(b * NHEADS + h) * HD * TLEN;
  const float*    rb = rel_bias + (size_t)h * TLEN * TLEN + (size_t)t0 * TLEN;

  const v16h qa = loadA_h(qp, HD, l, half, t0, 0);   // HEAD_DIM==32: one A fragment

  // All-ones B fragment: row sums of P via the matrix pipe (P x 1 -> each lane
  // of a half-group receives its row's sum in the C/D broadcast layout).
  v16h ones;
#pragma unroll
  for (int i = 0; i < 16; ++i) ones[i] = (_Float16)1.0f;

  v8f acc0 = {}; v8f acc1 = {};
  float mrun[8], lrun[8];
#pragma unroll
  for (int i = 0; i < 8; ++i) { mrun[i] = -3.0e38f; lrun[i] = 0.f; }

  _Float16* pw = &psh[wid][0];

  for (int s0 = 0; s0 < TLEN; s0 += 32) {
    if (s0 + 32 < TLEN) {   // hint next K chunk into cache (global_prefetch_b8)
      __builtin_prefetch(kp + (size_t)(s0 + 32 + l) * HD, 0, 1);
      __builtin_prefetch(kp + (size_t)(s0 + 48 + l) * HD, 0, 1);
    }
    v16h kb0 = loadB(kp, HD, l, half, s0,      0);
    v16h kb1 = loadB(kp, HD, l, half, s0 + 16, 0);
    v8f z = {};
    v8f st0 = __builtin_amdgcn_wmma_f32_16x16x32_f16(false, qa, false, kb0, (short)0, z, false, false);
    v8f st1 = __builtin_amdgcn_wmma_f32_16x16x32_f16(false, qa, false, kb1, (short)0, z, false, false);

#pragma unroll
    for (int i = 0; i < 8; ++i) {           // additive relative position bias
      const float* rr = rb + (size_t)(i + 8 * half) * TLEN + s0;
      st0[i] += rr[l];
      st1[i] += rr[16 + l];
    }

    float corr[8];
#pragma unroll
    for (int i = 0; i < 8; ++i) {           // online max + exp; P tile -> LDS
      float mx = rowmax16(fmaxf(st0[i], st1[i]));
      float mn = fmaxf(mrun[i], mx);
      corr[i] = __expf(mrun[i] - mn);
      mrun[i] = mn;
      float p0 = __expf(st0[i] - mn);
      float p1 = __expf(st1[i] - mn);
      int row = i + 8 * half;
      pw[row * 32 + l]      = (_Float16)p0;  // C-layout -> LDS tile [row][s]
      pw[row * 32 + 16 + l] = (_Float16)p1;
    }
    __builtin_amdgcn_wave_barrier();        // keep LDS RAW ordering (in-order per wave)
    const _Float16* pr = pw + l * 32 + half * 8;
    v8h pa = *(const v8h*)pr;
    v8h pb = *(const v8h*)(pr + 16);
    v16h pf = combine16(pa, pb);            // P as A-fragment

    v16h vb0 = loadB(vp, TLEN, l, half, 0,  s0);
    v16h vb1 = loadB(vp, TLEN, l, half, 16, s0);
    v8f rs = __builtin_amdgcn_wmma_f32_16x16x32_f16(false, pf, false, ones, (short)0, z, false, false);
#pragma unroll
    for (int i = 0; i < 8; ++i) {           // rescale running state on the VALU pipe
      lrun[i] = lrun[i] * corr[i] + rs[i];
      acc0[i] *= corr[i];
      acc1[i] *= corr[i];
    }
    acc0 = __builtin_amdgcn_wmma_f32_16x16x32_f16(false, pf, false, vb0, (short)0, acc0, false, false);
    acc1 = __builtin_amdgcn_wmma_f32_16x16x32_f16(false, pf, false, vb1, (short)0, acc1, false, false);
    __builtin_amdgcn_wave_barrier();
  }

  _Float16* cb = ctxh + (size_t)b * TLEN * DIM + h * HD;
#pragma unroll
  for (int i = 0; i < 8; ++i) {
    float inv = 1.0f / lrun[i];
    int t = t0 + i + 8 * half;
    cb[(size_t)t * DIM + l]      = (_Float16)(acc0[i] * inv);
    cb[(size_t)t * DIM + 16 + l] = (_Float16)(acc1[i] * inv);
  }
}

// ---------------- output projection ----------------
__global__ __launch_bounds__(256) void oproj_kernel(
    const _Float16* __restrict__ ctxh, const _Float16* __restrict__ WpH,
    const float* __restrict__ bp, float* __restrict__ out) {
  const int lane = threadIdx.x & 31;
  const int wid  = threadIdx.x >> 5;
  const int l = lane & 15, half = lane >> 4;
  const int wl = blockIdx.x * 8 + wid;       // 0..3455
  const int mt = wl >> 4;
  const int nt = wl & 15;
  const int m0 = mt * 16, n0 = nt * 32;

  v8f c0 = {}; v8f c1 = {};
#pragma unroll 4
  for (int k0 = 0; k0 < DIM; k0 += 32) {
    v16h a  = loadA_h(ctxh, DIM, l, half, m0, k0);
    v16h b0 = loadB(WpH, DIM, l, half, n0,      k0);
    v16h b1 = loadB(WpH, DIM, l, half, n0 + 16, k0);
    c0 = __builtin_amdgcn_wmma_f32_16x16x32_f16(false, a, false, b0, (short)0, c0, false, false);
    c1 = __builtin_amdgcn_wmma_f32_16x16x32_f16(false, a, false, b1, (short)0, c1, false, false);
  }
  const float b0v = bp[n0 + l], b1v = bp[n0 + 16 + l];
#pragma unroll
  for (int i = 0; i < 8; ++i) {
    int m = m0 + i + 8 * half;
    out[(size_t)m * DIM + n0 + l]      = c0[i] + b0v;
    out[(size_t)m * DIM + n0 + 16 + l] = c1[i] + b1v;
  }
}

extern "C" void kernel_launch(void* const* d_in, const int* in_sizes, int n_in,
                              void* d_out, int out_size, void* d_ws, size_t ws_size,
                              hipStream_t stream) {
  const float* query = (const float*)d_in[0];
  const float* key   = (const float*)d_in[1];
  const float* value = (const float*)d_in[2];
  const float* Wq = (const float*)d_in[3];
  const float* bq = (const float*)d_in[4];
  const float* Wk = (const float*)d_in[5];
  const float* bk = (const float*)d_in[6];
  const float* Wv = (const float*)d_in[7];
  const float* bv = (const float*)d_in[8];
  const float* Wp = (const float*)d_in[9];
  const float* bp = (const float*)d_in[10];
  const float* ls = (const float*)d_in[11];
  const float* rb = (const float*)d_in[12];

  char* ws = (char*)d_ws;
  const size_t WSZ   = (size_t)4 * DIM * DIM * sizeof(_Float16);   // 2 MB
  const size_t BHTD  = (size_t)BATCH * NHEADS * TLEN * HD;         // elements
  _Float16* WH   = (_Float16*)ws;
  _Float16* qh   = (_Float16*)(ws + WSZ);
  _Float16* kh   = qh + BHTD;
  _Float16* vT   = kh + BHTD;
  _Float16* ctxh = vT + BHTD;                                      // [B,T,512] f16

  cvt_w_kernel<<<4096, 256, 0, stream>>>(Wq, Wk, Wv, Wp, WH);
  qkv_proj_kernel<<<(3 * MTILES * NHEADS) / 8, 256, 0, stream>>>(
      query, key, value, WH, bq, bk, bv, ls, qh, kh, vT);
  dim3 ag((STILES + 7) / 8, BATCH * NHEADS);
  attn_kernel<<<ag, 256, 0, stream>>>(qh, kh, vT, rb, ctxh);
  oproj_kernel<<<(MTILES * NHEADS) / 8, 256, 0, stream>>>(
      ctxh, WH + (size_t)3 * DIM * DIM, bp, (float*)d_out);
}
